// PreferenceEmbedding_50886772523482
// MI455X (gfx1250) — matrix operand order repaired and checked by reading, not compiled, not run on hardware
//
#include <hip/hip_runtime.h>
#include <hip/hip_bf16.h>
#include <math.h>

// ---------------- types ----------------
typedef __attribute__((ext_vector_type(16))) _Float16 v16h;
typedef __attribute__((ext_vector_type(8)))  float    v8f;
typedef __attribute__((ext_vector_type(4)))  float    f32x4;
typedef __attribute__((ext_vector_type(4)))  _Float16 f16x4;

// ---------------- problem dims ----------------
#define NTOT   65536
#define XD     512
#define HD     1024
#define CD     64
#define CB     512
#define OD     64
#define TM     32          // rows per workgroup (2 M-tiles of 16)
#define NWAVE  8           // 256 threads / 32 lanes

// ---------------- LDS layout (bytes) ----------------
#define SM_ACTA   (TM*HD*2)
#define SM_ACTB   (TM*HD*2)
#define SM_ZF32   (TM*CD*4)
#define SM_ZF16   (TM*CD*2)
#define SM_ZQ     (TM*CD*2)
#define SM_AMIN   (TM*8)
#define SM_RED    (256*4)
#define SM_RED2   (32*4)
#define SM_TOTAL  (SM_ACTA+SM_ACTB+SM_ZF32+SM_ZF16+SM_ZQ+SM_AMIN+SM_RED+SM_RED2)

// ---------------- workspace layout (bytes) ----------------
#define OFF_W1T   0u                           // [1024][512]  f16
#define OFF_W2T   (OFF_W1T  + 1024u*512u*2u)   // [64][1024]   f16
#define OFF_EMBH  (OFF_W2T  + 64u*1024u*2u)    // [512][64]    f16
#define OFF_TE1T  (OFF_EMBH + 512u*64u*2u)     // [1024][64]   f16
#define OFF_TE2T  (OFF_TE1T + 1024u*64u*2u)    // [1024][1024] f16
#define OFF_MNT   (OFF_TE2T + 1024u*1024u*2u)  // [64][1024]   f16
#define OFF_LVT   (OFF_MNT  + 64u*1024u*2u)    // [64][1024]   f16
#define OFF_ENRM  (OFF_LVT  + 64u*1024u*2u)    // [512]        f32
#define OFF_PART  (OFF_ENRM + 512u*4u)         // [NTOT/TM]    f32

// ================= WMMA operand loaders =================
// A: 16x32 f16 tile from LDS, row-major, lda halfs.
// ISA: lane<16 -> M=lane, halves 0..7 = K+0..7, 8..15 = K+16..23
//      lane>=16 -> M=lane-16, halves 0..7 = K+8..15, 8..15 = K+24..31
__device__ __forceinline__ v16h load_a(const _Float16* a, int lda, int m0, int k0, int lane) {
  const int row = m0 + (lane & 15);
  const int kb  = k0 + ((lane >> 4) << 3);
  const _Float16* p = a + row * lda + kb;
  v16h r;
  *reinterpret_cast<f32x4*>(&r)       = *reinterpret_cast<const f32x4*>(p);
  *(reinterpret_cast<f32x4*>(&r) + 1) = *reinterpret_cast<const f32x4*>(p + 16);
  return r;
}

// B: 32x16 f16 tile from W^T [Nout][Kin] row-major in global (ldb=K halfs).
// ISA: lanes 0..15 -> N=lane, K=k0+0..7,16..23; lanes 16..31 -> N=lane-16, K=k0+8..15,24..31
__device__ __forceinline__ v16h load_b(const _Float16* b, int ldb, int n0, int k0, int lane) {
  const int col = n0 + (lane & 15);
  const int kb  = k0 + ((lane >> 4) << 3);
  const _Float16* p = b + (size_t)col * ldb + kb;
  v16h r;
  *reinterpret_cast<f32x4*>(&r)       = *reinterpret_cast<const f32x4*>(p);
  *(reinterpret_cast<f32x4*>(&r) + 1) = *reinterpret_cast<const f32x4*>(p + 16);
  return r;
}

// C/D: VGPR r, lanes 0..15 -> M=m0+r, N=n0+lane ; lanes 16..31 -> M=m0+8+r, N=n0+lane-16
__device__ __forceinline__ void store_tile_f16(_Float16* dst, int ldd, int m0, int n0,
                                               int lane, v8f acc, bool do_tanh) {
  const int mbase = m0 + ((lane >> 4) << 3);
  const int col   = n0 + (lane & 15);
#pragma unroll
  for (int r = 0; r < 8; ++r) {
    float v = acc[r];
    if (do_tanh) v = tanhf(v);
    dst[(mbase + r) * ldd + col] = (_Float16)v;
  }
}

// single-tile accumulate (stage 2)
__device__ __forceinline__ v8f wmma_acc(const _Float16* A, int lda, int m0,
                                        const _Float16* B, int ldb, int n0,
                                        int K, float bias, int lane) {
  v8f acc;
#pragma unroll
  for (int i = 0; i < 8; ++i) acc[i] = bias;
#pragma unroll 4
  for (int k0 = 0; k0 < K; k0 += 32) {
    v16h av = load_a(A, lda, m0, k0, lane);
    v16h bv = load_b(B, ldb, n0, k0, lane);
    acc = __builtin_amdgcn_wmma_f32_16x16x32_f16(false, av, false, bv,
                                                 (short)0, acc, false, false);
  }
  return acc;
}

// dual-M accumulate (stage 6): B loaded once per k-step, 2 WMMAs per B fetch
__device__ __forceinline__ void wmma_acc2(const _Float16* A, int lda,
                                          const _Float16* B, int ldb, int n0,
                                          int K, float bias, int lane,
                                          v8f& acc0, v8f& acc1) {
#pragma unroll
  for (int i = 0; i < 8; ++i) { acc0[i] = bias; acc1[i] = bias; }
#pragma unroll 4
  for (int k0 = 0; k0 < K; k0 += 32) {
    v16h bv = load_b(B, ldb, n0, k0, lane);
    v16h a0 = load_a(A, lda, 0,  k0, lane);
    v16h a1 = load_a(A, lda, 16, k0, lane);
    acc0 = __builtin_amdgcn_wmma_f32_16x16x32_f16(false, a0, false, bv,
                                                  (short)0, acc0, false, false);
    acc1 = __builtin_amdgcn_wmma_f32_16x16x32_f16(false, a1, false, bv,
                                                  (short)0, acc1, false, false);
  }
}

// 2x2 register block: 2 N-tiles x 2 M-tiles, 4 WMMAs per 8 mem ops (2 ops/WMMA)
__device__ __forceinline__ void wmma_acc4(const _Float16* A, int lda,
                                          const _Float16* B, int ldb,
                                          int n0, int n1, int K,
                                          float bv0, float bv1, int lane,
                                          v8f& c00, v8f& c01, v8f& c10, v8f& c11) {
#pragma unroll
  for (int i = 0; i < 8; ++i) { c00[i] = bv0; c10[i] = bv0; c01[i] = bv1; c11[i] = bv1; }
#pragma unroll 2
  for (int k0 = 0; k0 < K; k0 += 32) {
    v16h b0 = load_b(B, ldb, n0, k0, lane);
    v16h b1 = load_b(B, ldb, n1, k0, lane);
    v16h a0 = load_a(A, lda, 0,  k0, lane);
    v16h a1 = load_a(A, lda, 16, k0, lane);
    c00 = __builtin_amdgcn_wmma_f32_16x16x32_f16(false, a0, false, b0, (short)0, c00, false, false);
    c01 = __builtin_amdgcn_wmma_f32_16x16x32_f16(false, a0, false, b1, (short)0, c01, false, false);
    c10 = __builtin_amdgcn_wmma_f32_16x16x32_f16(false, a1, false, b0, (short)0, c10, false, false);
    c11 = __builtin_amdgcn_wmma_f32_16x16x32_f16(false, a1, false, b1, (short)0, c11, false, false);
  }
}

// generic GEMM stage (2x2 blocked): C[TM x 16*ntiles] -> LDS f16 (ntiles even)
__device__ __forceinline__ void gemm_stage(const _Float16* A, int lda,
                                           const _Float16* B, int ldb,
                                           const float* bias, int K, int ntiles,
                                           _Float16* dst, int ldd, bool do_tanh,
                                           int wave, int lane) {
  for (int nt = wave * 2; nt < ntiles; nt += NWAVE * 2) {
    const int n0 = nt * 16;
    const int n1 = n0 + 16;
    const float bv0 = bias[n0 + (lane & 15)];
    const float bv1 = bias[n1 + (lane & 15)];
    v8f c00, c01, c10, c11;
    wmma_acc4(A, lda, B, ldb, n0, n1, K, bv0, bv1, lane, c00, c01, c10, c11);
    store_tile_f16(dst, ldd, 0,  n0, lane, c00, do_tanh);
    store_tile_f16(dst, ldd, 0,  n1, lane, c01, do_tanh);
    store_tile_f16(dst, ldd, 16, n0, lane, c10, do_tanh);
    store_tile_f16(dst, ldd, 16, n1, lane, c11, do_tanh);
  }
}

// ================= prep kernels =================
__global__ void k_transpose_f16(const float* __restrict__ src, _Float16* __restrict__ dst,
                                int K, int Nn) {
  int i = blockIdx.x * blockDim.x + threadIdx.x;   // index over dst [Nn][K]
  if (i < K * Nn) {
    int n = i / K;
    int k = i - n * K;
    dst[i] = (_Float16)src[k * Nn + n];
  }
}

__global__ void k_convert_f16(const float* __restrict__ src, _Float16* __restrict__ dst, int total) {
  int i = blockIdx.x * blockDim.x + threadIdx.x;
  if (i < total) dst[i] = (_Float16)src[i];
}

__global__ void k_emb_norm(const float* __restrict__ emb, float* __restrict__ enorm) {
  int c = blockIdx.x * blockDim.x + threadIdx.x;
  if (c < CB) {
    float s = 0.f;
    for (int d = 0; d < CD; ++d) { float v = emb[c * CD + d]; s += v * v; }
    enorm[c] = s;
  }
}

// ================= argmin helper =================
__device__ __forceinline__ void argmin_update(unsigned long long* amin, int m,
                                              float dist, int code) {
  unsigned kb = __float_as_uint(dist);
  kb = (kb & 0x80000000u) ? ~kb : (kb | 0x80000000u);   // order-preserving map
  unsigned long long packed = ((unsigned long long)kb << 32) | (unsigned long long)code;
  atomicMin(&amin[m], packed);                          // ds_min_u64
}

// ================= fused megakernel =================
// __launch_bounds__(256, 4): 4 waves/SIMD -> caps VGPRs at 256/wave,
// no s_set_vgpr_msb traffic; 2 workgroups (2x145KB LDS) share a WGP.
__global__ __launch_bounds__(256, 4) void k_fused(
    const float* __restrict__ x,
    const float* __restrict__ vq_b1, const float* __restrict__ vq_b2,
    const float* __restrict__ emb,
    const float* __restrict__ te_b1, const float* __restrict__ te_b2,
    const float* __restrict__ mean_b, const float* __restrict__ lv_b,
    const _Float16* __restrict__ w1t, const _Float16* __restrict__ w2t,
    const _Float16* __restrict__ embh,
    const _Float16* __restrict__ te1t, const _Float16* __restrict__ te2t,
    const _Float16* __restrict__ meant, const _Float16* __restrict__ lvt,
    const float* __restrict__ enorm,
    float* __restrict__ out_mean, float* __restrict__ out_lv,
    float* __restrict__ partials) {
  extern __shared__ __align__(16) char smem[];
  _Float16* actA = (_Float16*)smem;
  _Float16* actB = (_Float16*)(smem + SM_ACTA);
  float*    zf32 = (float*)   (smem + SM_ACTA + SM_ACTB);
  _Float16* zf16 = (_Float16*)(smem + SM_ACTA + SM_ACTB + SM_ZF32);
  _Float16* zq   = (_Float16*)(smem + SM_ACTA + SM_ACTB + SM_ZF32 + SM_ZF16);
  unsigned long long* amin =
      (unsigned long long*)(smem + SM_ACTA + SM_ACTB + SM_ZF32 + SM_ZF16 + SM_ZQ);
  float* red  = (float*)(smem + SM_ACTA + SM_ACTB + SM_ZF32 + SM_ZF16 + SM_ZQ + SM_AMIN);
  float* red2 = (float*)(smem + SM_ACTA + SM_ACTB + SM_ZF32 + SM_ZF16 + SM_ZQ + SM_AMIN + SM_RED);

  const int tid  = threadIdx.x;
  const int lane = tid & 31;
  const int wave = tid >> 5;
  const int rowbase = blockIdx.x * TM;

  // ---- load x tile (TM x XD), f32 -> f16, into actA (lda = XD) ----
  const float* xg = x + (size_t)rowbase * XD;
  for (int i = tid * 4; i < TM * XD; i += 256 * 4) {
    f32x4 v = *reinterpret_cast<const f32x4*>(xg + i);
    f16x4 h4;
    h4[0] = (_Float16)v[0]; h4[1] = (_Float16)v[1];
    h4[2] = (_Float16)v[2]; h4[3] = (_Float16)v[3];
    *reinterpret_cast<f16x4*>(actA + i) = h4;
  }
  __syncthreads();

  // ---- stage 1: h = tanh(x @ W1 + b1)  [TM x HD] -> actB ----
  gemm_stage(actA, XD, w1t, XD, vq_b1, XD, HD / 16, actB, HD, true, wave, lane);
  __syncthreads();

  // ---- stage 2: z = h @ W2 + b2  [TM x CD] -> zf32 + zf16 ----
  // only 4 n-tiles: use (m,n) jobs so all 8 waves stay busy
  {
    const int njobs = 2 * (CD / 16);   // 8 jobs
    for (int job = wave; job < njobs; job += NWAVE) {
      const int m0 = (job & 1) * 16;
      const int n0 = (job >> 1) * 16;
      const float bv = vq_b2[n0 + (lane & 15)];
      v8f acc = wmma_acc(actB, HD, m0, w2t, HD, n0, HD, bv, lane);
      const int mbase = m0 + ((lane >> 4) << 3);
      const int col   = n0 + (lane & 15);
#pragma unroll
      for (int r = 0; r < 8; ++r) {
        zf32[(mbase + r) * CD + col] = acc[r];
        zf16[(mbase + r) * CD + col] = (_Float16)acc[r];
      }
    }
  }
  if (tid < TM) amin[tid] = ~0ull;
  __syncthreads();

  // ---- stage 3: scores = z @ emb^T ; argmin of (||e||^2 - 2*score) ----
  {
    for (int nt = wave * 2; nt < CB / 16; nt += NWAVE * 2) {
      const int n0 = nt * 16;
      const int n1 = n0 + 16;
      v8f c00, c01, c10, c11;
      wmma_acc4(zf16, CD, embh, CD, n0, n1, CD, 0.0f, 0.0f, lane, c00, c01, c10, c11);
      const int mhalf = (lane >> 4) << 3;
      const int code0 = n0 + (lane & 15);
      const int code1 = n1 + (lane & 15);
      const float en0 = enorm[code0];
      const float en1 = enorm[code1];
#pragma unroll
      for (int r = 0; r < 8; ++r) {
        argmin_update(amin, mhalf + r,      en0 - 2.0f * c00[r], code0);
        argmin_update(amin, mhalf + r,      en1 - 2.0f * c01[r], code1);
        argmin_update(amin, 16 + mhalf + r, en0 - 2.0f * c10[r], code0);
        argmin_update(amin, 16 + mhalf + r, en1 - 2.0f * c11[r], code1);
      }
    }
  }
  __syncthreads();

  // ---- gather z_q = emb[idx], accumulate sum((z_q - z)^2) ----
  {
    float lp = 0.f;
    for (int i = tid; i < TM * CD; i += 256) {
      const int r = i >> 6;               // CD == 64
      const int d = i & 63;
      const int idx = (int)(amin[r] & 0xffffffffu);
      const float e = emb[idx * CD + d];
      zq[i] = (_Float16)e;
      const float diff = e - zf32[i];
      lp += diff * diff;
    }
    red[tid] = lp;
    __syncthreads();
    if (tid < 32) {                       // fixed-order reduction (deterministic)
      float s = 0.f;
      for (int j = 0; j < 8; ++j) s += red[tid * 8 + j];
      red2[tid] = s;
    }
    __syncthreads();
    if (tid == 0) {
      float s = 0.f;
      for (int j = 0; j < 32; ++j) s += red2[j];
      partials[blockIdx.x] = s;
    }
  }
  __syncthreads();

  // ---- stage 4: t1 = tanh(z_q @ te_w1 + b)  [TM x HD] -> actA ----
  gemm_stage(zq, CD, te1t, CD, te_b1, CD, HD / 16, actA, HD, true, wave, lane);
  __syncthreads();

  // ---- stage 5: t2 = tanh(t1 @ te_w2 + b)  [TM x HD] -> actB ----
  gemm_stage(actA, HD, te2t, HD, te_b2, HD, HD / 16, actB, HD, true, wave, lane);
  __syncthreads();

  // ---- stage 6: mean / log_var -> global (8 column-jobs, one per wave) ----
  {
    const int mat = wave >> 2;            // 0 -> mean, 1 -> log_var
    const int n0  = (wave & 3) * 16;      // OD/16 == 4 column tiles
    const _Float16* B = mat ? lvt : meant;
    const float* bias = mat ? lv_b : mean_b;
    const float bv = bias[n0 + (lane & 15)];
    v8f acc0, acc1;
    wmma_acc2(actB, HD, B, HD, n0, HD, bv, lane, acc0, acc1);
    float* outp = mat ? out_lv : out_mean;
    const int mhalf = (lane >> 4) << 3;
    const int col   = n0 + (lane & 15);
#pragma unroll
    for (int r = 0; r < 8; ++r) {
      outp[(size_t)(rowbase + mhalf + r) * OD + col]      = acc0[r];
      outp[(size_t)(rowbase + 16 + mhalf + r) * OD + col] = acc1[r];
    }
  }
}

// ================= deterministic loss reduce =================
__global__ void k_loss_reduce(const float* __restrict__ partials, int n,
                              float* __restrict__ out) {
  __shared__ float red[256];
  const int tid = threadIdx.x;
  float s = 0.f;
  for (int i = tid; i < n; i += 256) s += partials[i];   // fixed per-thread order
  red[tid] = s;
  __syncthreads();
  if (tid == 0) {
    float t = 0.f;
    for (int j = 0; j < 256; ++j) t += red[j];           // fixed order
    // vq_loss + commitment = 2 * mean((z_q - z)^2)
    *out = t * (2.0f / ((float)NTOT * (float)CD));
  }
}

// ================= host entry =================
extern "C" void kernel_launch(void* const* d_in, const int* in_sizes, int n_in,
                              void* d_out, int out_size, void* d_ws, size_t ws_size,
                              hipStream_t stream) {
  const float* x      = (const float*)d_in[0];
  const float* vq_w1  = (const float*)d_in[1];
  const float* vq_b1  = (const float*)d_in[2];
  const float* vq_w2  = (const float*)d_in[3];
  const float* vq_b2  = (const float*)d_in[4];
  const float* emb    = (const float*)d_in[5];
  const float* te_w1  = (const float*)d_in[6];
  const float* te_b1  = (const float*)d_in[7];
  const float* te_w2  = (const float*)d_in[8];
  const float* te_b2  = (const float*)d_in[9];
  const float* mean_w = (const float*)d_in[10];
  const float* mean_b = (const float*)d_in[11];
  const float* lv_w   = (const float*)d_in[12];
  const float* lv_b   = (const float*)d_in[13];

  char* ws = (char*)d_ws;
  _Float16* w1t   = (_Float16*)(ws + OFF_W1T);
  _Float16* w2t   = (_Float16*)(ws + OFF_W2T);
  _Float16* embh  = (_Float16*)(ws + OFF_EMBH);
  _Float16* te1t  = (_Float16*)(ws + OFF_TE1T);
  _Float16* te2t  = (_Float16*)(ws + OFF_TE2T);
  _Float16* meant = (_Float16*)(ws + OFF_MNT);
  _Float16* lvt   = (_Float16*)(ws + OFF_LVT);
  float*    enorm = (float*)(ws + OFF_ENRM);
  float*    part  = (float*)(ws + OFF_PART);

  float* out_mean = (float*)d_out;
  float* out_lv   = out_mean + (size_t)NTOT * OD;
  float* out_loss = out_lv + (size_t)NTOT * OD;

  // ---- prep: transpose+convert weights to f16 W^T, emb f16, ||e||^2 ----
  k_transpose_f16<<<(XD * HD + 255) / 256, 256, 0, stream>>>(vq_w1, w1t, XD, HD);
  k_transpose_f16<<<(HD * CD + 255) / 256, 256, 0, stream>>>(vq_w2, w2t, HD, CD);
  k_convert_f16 <<<(CB * CD + 255) / 256, 256, 0, stream>>>(emb, embh, CB * CD);
  k_transpose_f16<<<(CD * HD + 255) / 256, 256, 0, stream>>>(te_w1, te1t, CD, HD);
  k_transpose_f16<<<(HD * HD + 255) / 256, 256, 0, stream>>>(te_w2, te2t, HD, HD);
  k_transpose_f16<<<(HD * OD + 255) / 256, 256, 0, stream>>>(mean_w, meant, HD, OD);
  k_transpose_f16<<<(HD * OD + 255) / 256, 256, 0, stream>>>(lv_w, lvt, HD, OD);
  k_emb_norm<<<(CB + 255) / 256, 256, 0, stream>>>(emb, enorm);

  // ---- fused megakernel: 2048 WGs x 256 threads, ~145 KB LDS (2 WGs/WGP) ----
  (void)hipFuncSetAttribute((const void*)k_fused,
                            hipFuncAttributeMaxDynamicSharedMemorySize, SM_TOTAL);
  k_fused<<<NTOT / TM, 256, SM_TOTAL, stream>>>(
      x, vq_b1, vq_b2, emb, te_b1, te_b2, mean_b, lv_b,
      w1t, w2t, embh, te1t, te2t, meant, lvt, enorm,
      out_mean, out_lv, part);

  // ---- deterministic loss reduction ----
  k_loss_reduce<<<1, 256, 0, stream>>>(part, NTOT / TM, out_loss);
}
